// SoftUnification_46179488367079
// MI455X (gfx1250) — compile-verified, matrix operand order checked by reading, not run on hardware
//
#include <hip/hip_runtime.h>
#include <hip/hip_bf16.h>
#include <stdint.h>

typedef __bf16 v16bf __attribute__((ext_vector_type(16)));
typedef float  v8f   __attribute__((ext_vector_type(8)));
typedef unsigned int u32x4 __attribute__((ext_vector_type(4)));
typedef int          i32x8 __attribute__((ext_vector_type(8)));
typedef int          i32x4 __attribute__((ext_vector_type(4)));

#define DSYM 64
#define NCAND 2048
#define TINV 10.0f   // 1 / temperature
#define TSTR 68      // padded LDS tile stride (floats) == 64 + TDM pad of 4 DW

__device__ __forceinline__ float gelu_exact(float x) {
    // exact gelu: 0.5*x*(1+erf(x/sqrt(2)))
    return 0.5f * x * (1.0f + erff(x * 0.70710678118654752f));
}

// Tensor Data Mover: DMA one 16x64 fp32 tile (row-major, row stride 64 floats)
// from global memory into LDS at lds_off, inserting 4 DWORDs of padding after
// every 64 DWORDs (one row) -> LDS row stride = 68 floats (bank-conflict-free
// for the 16-row column gathers below). Tracked on TENSORcnt.
__device__ __forceinline__ void tdm_load_tile(unsigned lds_off, const float* gptr) {
    const unsigned long long ga = (unsigned long long)(uintptr_t)gptr;
    const unsigned ga_lo = __builtin_amdgcn_readfirstlane((unsigned)ga);
    const unsigned ga_hi = __builtin_amdgcn_readfirstlane((unsigned)(ga >> 32));
    const unsigned loff  = __builtin_amdgcn_readfirstlane(lds_off);

    u32x4 g0;
    g0[0] = 1u;                                     // count=1, user D#, no gather
    g0[1] = loff;                                   // lds_addr (bytes)
    g0[2] = ga_lo;                                  // global_addr[31:0]
    g0[3] = (ga_hi & 0x01FFFFFFu) | (2u << 30);     // global_addr[56:32] | type=2

    i32x8 g1;
    g1[0] = (int)((2u << 16)     // data_size = 4 bytes
                | (1u << 20)     // pad_enable
                | (5u << 22)     // pad_interval: 64 DWORDs
                | (3u << 25));   // pad_amount:   4 DWORDs
    g1[1] = (int)(64u << 16);    // tensor_dim0 = 64 (low 16 bits in [63:48])
    g1[2] = (int)(16u << 16);    // tensor_dim1 = 16 (low 16 bits in [95:80])
    g1[3] = (int)(64u << 16);    // tile_dim0 = 64   ([127:112])
    g1[4] = 16;                  // tile_dim1 = 16   ([143:128]); tile_dim2 = 0
    g1[5] = 64;                  // tensor_dim0_stride = 64 ([191:160])
    g1[6] = (int)(1024u << 16);  // tensor_dim1_stride = 1024 (low bits [223:208])
    g1[7] = 0;

    const i32x4 z4 = {0, 0, 0, 0};                  // 2-D tensor: groups 2/3 unused
    const i32x8 z8 = {0, 0, 0, 0, 0, 0, 0, 0};      // trailing group (this toolchain)
    __builtin_amdgcn_tensor_load_to_lds(g0, g1, z4, z4, z8, 0);
}

__launch_bounds__(256)
__global__ void soft_unification_kernel(const float* __restrict__ query,
                                        const float* __restrict__ cand,
                                        const float* __restrict__ W1,
                                        const float* __restrict__ b1,
                                        const float* __restrict__ W2,
                                        const float* __restrict__ b2,
                                        float* __restrict__ out_w,
                                        float* __restrict__ out_u,
                                        int N) {
    __shared__ float qp[DSYM];                 // projected query + b1
    __shared__ float tiles[8][2][16 * TSTR];   // per-wave double-buffered tiles
    __shared__ float scores[NCAND];            // scores -> later reused as weights
    __shared__ float red[4][DSYM];             // partial weighted sums
    __shared__ float rbuf[8];                  // per-wave reduction buffer
    __shared__ float sb[2];                    // broadcast max / sum

    const int b    = blockIdx.x;
    const int t    = threadIdx.x;
    const int lane = t & 31;
    const int wave = t >> 5;
    const int hh   = lane >> 4;                // lane-half (0/1)
    const int l15  = lane & 15;

    const float* candB = cand + (size_t)b * N * DSYM;

    // ---- Stage 0: qp[d] = query[b] @ Wq + b1 -------------------------------
    if (t < DSYM) {
        float acc = b1[t];
        const float* qrow = query + b * DSYM;
        #pragma unroll 8
        for (int k = 0; k < DSYM; ++k)
            acc += qrow[k] * W1[k * DSYM + t];
        qp[t] = acc;
    }
    __syncthreads();

    // ---- Per-lane loop invariants ------------------------------------------
    float qv[4], w2v[4];
    #pragma unroll
    for (int nb = 0; nb < 4; ++nb) {
        qv[nb]  = qp[nb * 16 + l15];
        w2v[nb] = W2[nb * 16 + l15];
    }
    const float b2s = b2[0];

    // ---- Build resident Wc B-fragments (bf16 hi/lo split) ------------------
    // B-matrix (32x16) 16-bit layout: lane = N column, VGPR r + lane-half give
    // K pair: k = (r>>2)*16 + half*8 + (r&3)*2  (+ kk2*32 for the K step)
    v16bf Bh[2][4], Bl[2][4];
    #pragma unroll
    for (int kk2 = 0; kk2 < 2; ++kk2) {
        #pragma unroll
        for (int nb = 0; nb < 4; ++nb) {
            #pragma unroll
            for (int r = 0; r < 8; ++r) {
                const int k = kk2 * 32 + ((r >> 2) << 4) + (hh << 3) + ((r & 3) << 1);
                const int e = nb * 16 + l15;
                const float f0 = W1[(DSYM + k) * DSYM + e];      // Wc[k][e]
                const float f1 = W1[(DSYM + k + 1) * DSYM + e];  // Wc[k+1][e]
                const __bf16 h0 = (__bf16)f0;
                const __bf16 h1 = (__bf16)f1;
                Bh[kk2][nb][2 * r]     = h0;
                Bh[kk2][nb][2 * r + 1] = h1;
                Bl[kk2][nb][2 * r]     = (__bf16)(f0 - (float)h0);
                Bl[kk2][nb][2 * r + 1] = (__bf16)(f1 - (float)h1);
            }
        }
    }

    // ---- Pass 1: scores for all N candidates (TDM double-buffered) ---------
    float* tile0 = tiles[wave][0];
    float* tile1 = tiles[wave][1];
    const unsigned o0 = (unsigned)(uintptr_t)tile0;  // flat addr low 32b = LDS offset
    const unsigned o1 = (unsigned)(uintptr_t)tile1;

    const int chunks = N >> 7;                 // chunks-of-16 per wave (8 waves)
    const int nwave0 = wave * (N >> 3);

    tdm_load_tile(o0, candB + (size_t)nwave0 * DSYM);   // prefetch chunk 0

    for (int i = 0; i < chunks; ++i) {
        const int n0 = nwave0 + i * 16;

        if (i + 1 < chunks) {
            // LDS reads of buffer (i+1)&1 from iteration i-1 are already
            // complete (their data was consumed by that iteration's WMMAs).
            asm volatile("" ::: "memory");
            tdm_load_tile((i & 1) ? o0 : o1, candB + (size_t)(n0 + 16) * DSYM);
            __builtin_amdgcn_s_wait_tensorcnt(1);   // chunk i has landed
        } else {
            __builtin_amdgcn_s_wait_tensorcnt(0);   // last chunk has landed
        }
        asm volatile("" ::: "memory");
        float* tile = (i & 1) ? tile1 : tile0;

        v8f c0 = {}, c1 = {}, c2 = {}, c3 = {};
        #pragma unroll
        for (int kk2 = 0; kk2 < 2; ++kk2) {
            // A-matrix (16x32) 16-bit layout: lane = M row (l15),
            // k = kk2*32 + (r>>2)*16 + half*8 + (r&3)*2
            v16bf Ah, Al;
            #pragma unroll
            for (int r = 0; r < 8; ++r) {
                const int k = kk2 * 32 + ((r >> 2) << 4) + (hh << 3) + ((r & 3) << 1);
                const float2 f = *(const float2*)(tile + l15 * TSTR + k);
                const __bf16 h0 = (__bf16)f.x;
                const __bf16 h1 = (__bf16)f.y;
                Ah[2 * r]     = h0;
                Ah[2 * r + 1] = h1;
                Al[2 * r]     = (__bf16)(f.x - (float)h0);
                Al[2 * r + 1] = (__bf16)(f.y - (float)h1);
            }
            // split-precision accumulate: Ah*Bh + Ah*Bl + Al*Bh
            c0 = __builtin_amdgcn_wmma_f32_16x16x32_bf16(false, Al, false, Bh[kk2][0], (short)0, c0, false, false);
            c0 = __builtin_amdgcn_wmma_f32_16x16x32_bf16(false, Ah, false, Bl[kk2][0], (short)0, c0, false, false);
            c0 = __builtin_amdgcn_wmma_f32_16x16x32_bf16(false, Ah, false, Bh[kk2][0], (short)0, c0, false, false);
            c1 = __builtin_amdgcn_wmma_f32_16x16x32_bf16(false, Al, false, Bh[kk2][1], (short)0, c1, false, false);
            c1 = __builtin_amdgcn_wmma_f32_16x16x32_bf16(false, Ah, false, Bl[kk2][1], (short)0, c1, false, false);
            c1 = __builtin_amdgcn_wmma_f32_16x16x32_bf16(false, Ah, false, Bh[kk2][1], (short)0, c1, false, false);
            c2 = __builtin_amdgcn_wmma_f32_16x16x32_bf16(false, Al, false, Bh[kk2][2], (short)0, c2, false, false);
            c2 = __builtin_amdgcn_wmma_f32_16x16x32_bf16(false, Ah, false, Bl[kk2][2], (short)0, c2, false, false);
            c2 = __builtin_amdgcn_wmma_f32_16x16x32_bf16(false, Ah, false, Bh[kk2][2], (short)0, c2, false, false);
            c3 = __builtin_amdgcn_wmma_f32_16x16x32_bf16(false, Al, false, Bh[kk2][3], (short)0, c3, false, false);
            c3 = __builtin_amdgcn_wmma_f32_16x16x32_bf16(false, Ah, false, Bl[kk2][3], (short)0, c3, false, false);
            c3 = __builtin_amdgcn_wmma_f32_16x16x32_bf16(false, Ah, false, Bh[kk2][3], (short)0, c3, false, false);
        }

        // score_n = sum_d gelu(qp[d] + cp[n][d]) * W2[d] + b2
        // C layout: col d = nb*16 + l15; row n = n0 + r + 8*half
        #pragma unroll
        for (int r = 0; r < 8; ++r) {
            float g = gelu_exact(qv[0] + c0[r]) * w2v[0]
                    + gelu_exact(qv[1] + c1[r]) * w2v[1]
                    + gelu_exact(qv[2] + c2[r]) * w2v[2]
                    + gelu_exact(qv[3] + c3[r]) * w2v[3];
            g += __shfl_xor(g, 1);
            g += __shfl_xor(g, 2);
            g += __shfl_xor(g, 4);
            g += __shfl_xor(g, 8);
            if (l15 == 0)
                scores[n0 + hh * 8 + r] = g + b2s;
        }
    }
    __syncthreads();

    // ---- Pass 2a: softmax over N (temperature 0.1) -------------------------
    float m = -INFINITY;
    for (int j = t; j < N; j += 256) m = fmaxf(m, scores[j]);
    #pragma unroll
    for (int off = 16; off >= 1; off >>= 1) m = fmaxf(m, __shfl_xor(m, off));
    if (lane == 0) rbuf[wave] = m;
    __syncthreads();
    if (t == 0) {
        float mm = rbuf[0];
        #pragma unroll
        for (int i = 1; i < 8; ++i) mm = fmaxf(mm, rbuf[i]);
        sb[0] = mm;
    }
    __syncthreads();
    m = sb[0];

    float s = 0.0f;
    for (int j = t; j < N; j += 256) s += __expf((scores[j] - m) * TINV);
    #pragma unroll
    for (int off = 16; off >= 1; off >>= 1) s += __shfl_xor(s, off);
    if (lane == 0) rbuf[wave] = s;
    __syncthreads();
    if (t == 0) {
        float ss = rbuf[0];
        #pragma unroll
        for (int i = 1; i < 8; ++i) ss += rbuf[i];
        sb[1] = ss;
    }
    __syncthreads();
    const float invS = 1.0f / sb[1];

    for (int j = t; j < N; j += 256) {
        const float w = __expf((scores[j] - m) * TINV) * invS;
        out_w[(size_t)b * N + j] = w;
        scores[j] = w;                       // reuse LDS as weights
    }
    __syncthreads();

    // ---- Pass 2b: unified[b][d] = sum_n w[n] * cand[b][n][d] ---------------
    const int d   = t & 63;
    const int grp = t >> 6;                  // 4 n-groups
    float acc = 0.0f;
    for (int n = grp; n < N; n += 4)
        acc += scores[n] * candB[(size_t)n * DSYM + d];
    red[grp][d] = acc;
    __syncthreads();
    if (t < DSYM)
        out_u[(size_t)b * DSYM + t] = red[0][t] + red[1][t] + red[2][t] + red[3][t];
}

extern "C" void kernel_launch(void* const* d_in, const int* in_sizes, int n_in,
                              void* d_out, int out_size, void* d_ws, size_t ws_size,
                              hipStream_t stream) {
    const float* query = (const float*)d_in[0];
    const float* cand  = (const float*)d_in[1];
    const float* W1    = (const float*)d_in[2];
    const float* b1    = (const float*)d_in[3];
    const float* W2    = (const float*)d_in[4];
    const float* b2    = (const float*)d_in[5];

    const int B = in_sizes[0] / DSYM;                 // query is (B, 64)
    const int N = in_sizes[1] / in_sizes[0];          // (B*N*D)/(B*D)

    float* out_w = (float*)d_out;                     // weights: (B, N)
    float* out_u = out_w + (size_t)B * N;             // unified: (B, 64)

    soft_unification_kernel<<<dim3(B), dim3(256), 0, stream>>>(
        query, cand, W1, b1, W2, b2, out_w, out_u, N);
}
